// FlashBlockSparseAttention_77300821393856
// MI455X (gfx1250) — matrix operand order repaired and checked by reading, not compiled, not run on hardware
//
#include <hip/hip_runtime.h>
#include <hip/hip_bf16.h>

// ---------------------------------------------------------------------------
// FlashBlockSparseAttention for MI455X (gfx1250), wave32 + WMMA f16 path,
// with TDM (tensor_load_to_lds) + async-to-LDS staging.
// D=512, H=8, HD=64, BS=64, S=8192, NB=128.
// ---------------------------------------------------------------------------

typedef __attribute__((ext_vector_type(16))) _Float16 v16h;
typedef __attribute__((ext_vector_type(8)))  float    v8f;

#define WMMA_F16(a, b, c) \
    __builtin_amdgcn_wmma_f32_16x16x32_f16(false, (a), false, (b), (short)0, (c), false, false)

#ifndef ATHENA_TRY_TDM
#define ATHENA_TRY_TDM 1
#endif

#if defined(__has_builtin)
# if __has_builtin(__builtin_amdgcn_global_load_async_to_lds_b128)
#  define HAVE_ASYNC_LDS 1
# else
#  define HAVE_ASYNC_LDS 0
# endif
# if ATHENA_TRY_TDM && __has_builtin(__builtin_amdgcn_tensor_load_to_lds) && \
     __has_builtin(__builtin_amdgcn_s_wait_tensorcnt)
#  define HAVE_TDM 1
# else
#  define HAVE_TDM 0
# endif
#else
# define HAVE_ASYNC_LDS 0
# define HAVE_TDM 0
#endif

// Vector types matching the builtin signatures (GCC vector_size style, as
// printed by the clang diagnostics).
typedef int          av4i  __attribute__((vector_size(16)));
typedef unsigned int u32x4 __attribute__((vector_size(16)));
typedef int          i32x4 __attribute__((vector_size(16)));
typedef int          i32x8 __attribute__((vector_size(32)));

// Low 32 bits of a generic pointer to a __shared__ object are the LDS offset
// (ISA 10.2).
__device__ __forceinline__ unsigned lds_offset(const void* p) {
    return (unsigned)(size_t)p;
}

#if HAVE_ASYNC_LDS
__device__ __forceinline__ void async_b128(const _Float16* g, _Float16* l) {
    __builtin_amdgcn_global_load_async_to_lds_b128(
        (av4i*)(void*)g, (av4i*)(void*)l, 0, 0);
}
__device__ __forceinline__ void wait_async0() {
#if defined(__has_builtin) && __has_builtin(__builtin_amdgcn_s_wait_asynccnt)
    __builtin_amdgcn_s_wait_asynccnt(0);
#else
    asm volatile("s_wait_asynccnt 0" ::: "memory");
#endif
}
#endif

#if HAVE_TDM
// 2D tile load via Tensor Data Mover: tile_h rows of tile_w f16 elements,
// source row stride src_stride (elements), packed contiguously into LDS.
// Descriptor bitfields per CDNA5 ISA §8 (D# groups). Issue from ONE wave.
__device__ __forceinline__ void tdm_load_2d(const _Float16* gsrc, _Float16* ldst,
                                            unsigned tile_w, unsigned tile_h,
                                            unsigned src_stride) {
    unsigned long long ga = (unsigned long long)(size_t)gsrc;
    u32x4 g0 = {};
    g0[0] = 1u;                                    // count=1 (valid), user mode
    g0[1] = lds_offset(ldst);                      // lds_addr (bytes)
    g0[2] = (unsigned)ga;                          // global_addr[31:0]
    g0[3] = (unsigned)((ga >> 32) & 0x01ffffffu)   // global_addr[56:32]
            | (2u << 30);                          // type=2 ("image")
    i32x8 g1 = {};
    g1[0] = 0x00010000;                            // data_size=1 -> 2 bytes
    g1[1] = (int)((src_stride & 0xffffu) << 16);   // tensor_dim0[15:0]
    g1[2] = (int)(((src_stride >> 16) & 0xffffu)   // tensor_dim0[31:16]
            | ((tile_h & 0xffffu) << 16));         // tensor_dim1[15:0]
    g1[3] = (int)(((tile_h >> 16) & 0xffffu)       // tensor_dim1[31:16]
            | ((tile_w & 0xffffu) << 16));         // tile_dim0
    g1[4] = (int)(tile_h & 0xffffu);               // tile_dim1 (tile_dim2=0)
    g1[5] = (int)src_stride;                       // tensor_dim0_stride[31:0]
    g1[6] = 0;                                     // stride0 hi / dim1_stride lo
    g1[7] = 0;
    i32x4 g2 = {};
    g2[0] = 1;                                     // tensor_dim2 = 1
    i32x4 g3 = {};
    i32x8 g4 = {};                                 // 5th group (clang-23 arity)
    __builtin_amdgcn_tensor_load_to_lds(g0, g1, g2, g3, g4, 0);
}
__device__ __forceinline__ void wait_tensor0() {
    __builtin_amdgcn_s_wait_tensorcnt(0);
}
#endif

// ------------------------- fragment gather helpers -------------------------
// A-matrix 16x32 f16 fragment from row-major LDS tile (stride in halves).
// Lanes 0-15 hold M=lane, K in {0..15}; lanes 16-31 hold M=lane-16,
// K in {16..31}. VGPR v holds K = 2v (+16 for lanes 16-31).
__device__ __forceinline__ v16h load_a_frag(const _Float16* lds, int row_base,
                                            int k_base, int stride, int lane) {
    const int half_id = lane >> 4;
    const int m       = lane & 15;
    const _Float16* rowp = lds + (size_t)(row_base + m) * stride + k_base;
    v16h a;
#pragma unroll
    for (int v = 0; v < 8; ++v) {
        int k0 = 2 * v + half_id * 16;
        a[2 * v]     = rowp[k0];
        a[2 * v + 1] = rowp[k0 + 1];
    }
    return a;
}

// B-matrix 32x16 f16 fragment: lanes 0-15 hold N=lane, K=k_base..+15; lanes
// 16-31 hold N=lane-16, K=+16..+31. LDS tile stored [N][K] row-major.
__device__ __forceinline__ v16h load_b_frag(const _Float16* lds, int n_base,
                                            int k_base, int stride, int lane) {
    const int half_id = lane >> 4;
    const int n       = lane & 15;
    return *(const v16h*)(lds + (size_t)(n_base + n) * stride + k_base + half_id * 16);
}

// ------------------------------ f32 -> f16 ---------------------------------
__global__ void cvt_f32_to_f16(const float* __restrict__ in,
                               _Float16* __restrict__ out, int n) {
    int idx = blockIdx.x * 256 + threadIdx.x;
    if (idx < n) out[idx] = (_Float16)in[idx];
}

// --------------------------- QKV projection GEMM ---------------------------
// C[8192,1536] = x_f16[8192,512] @ w_qkv_f16[512,1536] + b_qkv
__global__ __launch_bounds__(128) void qkv_gemm_kernel(
    const _Float16* __restrict__ A, const _Float16* __restrict__ B,
    const float* __restrict__ bias,
    _Float16* __restrict__ qb, _Float16* __restrict__ kb, _Float16* __restrict__ vb) {
    __shared__ __align__(128) _Float16 sA[64 * 64];   // [m][k]
    __shared__ __align__(128) _Float16 sBT[64 * 64];  // [n][k]

    const int tid = threadIdx.x, lane = tid & 31, w = tid >> 5;
    const int row0 = blockIdx.y * 64, col0 = blockIdx.x * 64;
    const int half_id = lane >> 4, ln = lane & 15;

    v8f acc[4] = {};

    for (int ks = 0; ks < 512; ks += 64) {
        __syncthreads();
#if HAVE_ASYNC_LDS
        for (int t = tid; t < 512; t += 128) {
            int r = t >> 3, seg = t & 7;
            async_b128(&A[(size_t)(row0 + r) * 512 + ks + seg * 8],
                       &sA[r * 64 + seg * 8]);
        }
#else
        for (int t = tid; t < 512; t += 128) {
            int r = t >> 3, seg = t & 7;
            *(uint4*)&sA[r * 64 + seg * 8] =
                *(const uint4*)&A[(size_t)(row0 + r) * 512 + ks + seg * 8];
        }
#endif
        for (int t = tid; t < 512; t += 128) {
            int k = t >> 3, seg = t & 7;
            union { uint4 u; _Float16 h[8]; } cv;
            cv.u = *(const uint4*)&B[(size_t)(ks + k) * 1536 + col0 + seg * 8];
#pragma unroll
            for (int e = 0; e < 8; ++e) sBT[(seg * 8 + e) * 64 + k] = cv.h[e];
        }
#if HAVE_ASYNC_LDS
        wait_async0();
#endif
        __syncthreads();

        v16h a0 = load_a_frag(sA, w * 16, 0, 64, lane);
        v16h a1 = load_a_frag(sA, w * 16, 32, 64, lane);
#pragma unroll
        for (int n = 0; n < 4; ++n) {
            v16h b0 = load_b_frag(sBT, n * 16, 0, 64, lane);
            v16h b1 = load_b_frag(sBT, n * 16, 32, 64, lane);
            acc[n] = WMMA_F16(a0, b0, acc[n]);
            acc[n] = WMMA_F16(a1, b1, acc[n]);
        }
    }

    // Epilogue: scatter into q/k/v [H=8][S=8192][HD=64] f16 layout.
#pragma unroll
    for (int n = 0; n < 4; ++n) {
#pragma unroll
        for (int r = 0; r < 8; ++r) {
            int row = row0 + w * 16 + r + 8 * half_id;
            int col = col0 + n * 16 + ln;
            float val = acc[n][r] + bias[col];
            int t  = col >> 9;          // 0=q,1=k,2=v
            int rm = col & 511;
            int hh = rm >> 6, hd = rm & 63;
            _Float16* dst = (t == 0) ? qb : (t == 1) ? kb : vb;
            dst[((size_t)hh * 8192 + row) * 64 + hd] = (_Float16)val;
        }
    }
}

// --------------------------- router: block means ---------------------------
__global__ __launch_bounds__(512) void rep_kernel(const _Float16* __restrict__ q,
                                                  float* __restrict__ rep) {
    int nb = blockIdx.x, c = threadIdx.x;        // c in [0,512)
    int h = c >> 6, hd = c & 63;
    const _Float16* base = q + ((size_t)h * 8192 + nb * 64) * 64 + hd;
    float s = 0.f;
#pragma unroll 8
    for (int r = 0; r < 64; ++r) s += (float)base[r * 64];
    rep[nb * 512 + c] = s * (1.f / 64.f);
}

// ----------------------------- router: MLP ---------------------------------
__global__ __launch_bounds__(128) void router_kernel(
    const float* __restrict__ rep, const float* __restrict__ w_r1,
    const float* __restrict__ b_r1, const float* __restrict__ w_r2,
    const float* __restrict__ b_r2, float* __restrict__ keep) {
    int nb = blockIdx.x, c = threadIdx.x;        // c in [0,128)
    const float* r = rep + nb * 512;
    float acc = b_r1[c];
    for (int d = 0; d < 512; ++d) acc = fmaf(r[d], w_r1[d * 128 + c], acc);
    acc = fmaxf(acc, 0.f) * w_r2[c];
    __shared__ float red[128];
    red[c] = acc;
    __syncthreads();
    for (int s = 64; s > 0; s >>= 1) {
        if (c < s) red[c] += red[c + s];
        __syncthreads();
    }
    if (c == 0) {
        float imp = 1.f / (1.f + __expf(-(red[0] + b_r2[0])));
        keep[nb] = (imp >= 0.3f) ? 1.f : 0.f;
    }
}

// ------------------------------ attention ----------------------------------
__global__ __launch_bounds__(128) void attn_kernel(
    const _Float16* __restrict__ q, const _Float16* __restrict__ k,
    const _Float16* __restrict__ v, const float* __restrict__ keep,
    _Float16* __restrict__ attn) {
    __shared__ __align__(128) _Float16 sQ[64 * 64];   // [qrow][hd]
    __shared__ __align__(128) _Float16 sK[64 * 64];   // [key][hd]
    __shared__ __align__(128) _Float16 sVT[64 * 64];  // [hd][key]
    __shared__ __align__(128) _Float16 sP[64 * 64];   // [qrow][key]

    const int tid = threadIdx.x, lane = tid & 31, w = tid >> 5;
    const int h = blockIdx.x >> 7, i = blockIdx.x & 127;
    const int half_id = lane >> 4, ln = lane & 15;

    const float keep_i = keep[i];
    if (keep_i == 0.f) {   // whole output block gated to zero
        for (int t = tid; t < 512; t += 128) {
            int r = t >> 3, seg = t & 7;
            uint4 z = {0u, 0u, 0u, 0u};
            *(uint4*)&attn[(size_t)(i * 64 + r) * 512 + h * 64 + seg * 8] = z;
        }
        return;
    }

    const _Float16* qh = q + ((size_t)h * 8192 + i * 64) * 64;
#if HAVE_TDM
    if (w == 0) tdm_load_2d(qh, sQ, 4096u, 1u, 4096u);
    wait_tensor0();
#elif HAVE_ASYNC_LDS
    for (int t = tid; t < 512; t += 128) async_b128(qh + t * 8, sQ + t * 8);
    wait_async0();
#else
    for (int t = tid; t < 512; t += 128) {
        int r = t >> 3, seg = t & 7;
        *(uint4*)&sQ[r * 64 + seg * 8] = *(const uint4*)&qh[r * 64 + seg * 8];
    }
#endif
    __syncthreads();
    const v16h qa0 = load_a_frag(sQ, w * 16, 0, 64, lane);
    const v16h qa1 = load_a_frag(sQ, w * 16, 32, 64, lane);

    v8f oacc[4] = {};
    const float scale = 0.125f;   // 1/sqrt(64)

    for (int j = 0; j < 128; ++j) {
        if (keep[j] == 0.f) continue;       // block-sparse skip (uniform)
        __syncthreads();                    // previous tiles fully consumed

        const _Float16* kh = k + ((size_t)h * 8192 + j * 64) * 64;
        const _Float16* vh = v + ((size_t)h * 8192 + j * 64) * 64;
        if (j + 1 < 128)
            __builtin_prefetch(k + ((size_t)h * 8192 + (j + 1) * 64) * 64, 0, 3);

        // K tile: contiguous 8KB copy (TDM / async / manual)
#if HAVE_TDM
        if (w == 0) tdm_load_2d(kh, sK, 4096u, 1u, 4096u);
#elif HAVE_ASYNC_LDS
        for (int t = tid; t < 512; t += 128) async_b128(kh + t * 8, sK + t * 8);
#else
        for (int t = tid; t < 512; t += 128) {
            int r = t >> 3, seg = t & 7;
            *(uint4*)&sK[r * 64 + seg * 8] = *(const uint4*)&kh[r * 64 + seg * 8];
        }
#endif
        // V tile: manual transpose scatter ([key][hd] -> [hd][key])
        for (int t = tid; t < 512; t += 128) {
            int r = t >> 3, seg = t & 7;
            union { uint4 u; _Float16 h[8]; } cv;
            cv.u = *(const uint4*)&vh[r * 64 + seg * 8];
#pragma unroll
            for (int e = 0; e < 8; ++e) sVT[(seg * 8 + e) * 64 + r] = cv.h[e];
        }
#if HAVE_TDM
        wait_tensor0();
#elif HAVE_ASYNC_LDS
        wait_async0();
#endif
        __syncthreads();

        // S = q_i @ k_j^T : B[K=hd][N=key] gathered from sK[key][hd]
        v8f sacc[4] = {};
#pragma unroll
        for (int n = 0; n < 4; ++n) {
            v16h b0 = load_b_frag(sK, n * 16, 0, 64, lane);
            v16h b1 = load_b_frag(sK, n * 16, 32, 64, lane);
            sacc[n] = WMMA_F16(qa0, b0, sacc[n]);
            sacc[n] = WMMA_F16(qa1, b1, sacc[n]);
        }

        // scale + diagonal-block causal mask (strict upper triangle only)
#pragma unroll
        for (int n = 0; n < 4; ++n) {
#pragma unroll
            for (int r = 0; r < 8; ++r) {
                float s = sacc[n][r] * scale;
                if (j == i) {
                    int qrow = w * 16 + r + 8 * half_id;
                    int kcol = n * 16 + ln;
                    if (kcol > qrow) s = -1e30f;
                }
                sacc[n][r] = s;
            }
        }

        // per-row softmax over this key block (64 cols = 16 lanes x 4 tiles)
#pragma unroll
        for (int r = 0; r < 8; ++r) {
            float m = fmaxf(fmaxf(sacc[0][r], sacc[1][r]),
                            fmaxf(sacc[2][r], sacc[3][r]));
#pragma unroll
            for (int msk = 1; msk < 16; msk <<= 1) m = fmaxf(m, __shfl_xor(m, msk, 32));
            float sum = 0.f;
#pragma unroll
            for (int n = 0; n < 4; ++n) {
                float e = __expf(sacc[n][r] - m);
                sacc[n][r] = e;
                sum += e;
            }
#pragma unroll
            for (int msk = 1; msk < 16; msk <<= 1) sum += __shfl_xor(sum, msk, 32);
            float inv = keep_i * keep[j] / sum;   // gate folded into P
#pragma unroll
            for (int n = 0; n < 4; ++n) sacc[n][r] *= inv;
        }

        // stage P (per-wave private rows -> A layout re-gather after barrier)
#pragma unroll
        for (int n = 0; n < 4; ++n)
#pragma unroll
            for (int r = 0; r < 8; ++r)
                sP[(w * 16 + r + 8 * half_id) * 64 + n * 16 + ln] =
                    (_Float16)sacc[n][r];
        __syncthreads();

        v16h pa0 = load_a_frag(sP, w * 16, 0, 64, lane);
        v16h pa1 = load_a_frag(sP, w * 16, 32, 64, lane);
#pragma unroll
        for (int n = 0; n < 4; ++n) {
            v16h b0 = load_b_frag(sVT, n * 16, 0, 64, lane);
            v16h b1 = load_b_frag(sVT, n * 16, 32, 64, lane);
            oacc[n] = WMMA_F16(pa0, b0, oacc[n]);
            oacc[n] = WMMA_F16(pa1, b1, oacc[n]);
        }
    }

    // store O into attn[S, D] (col = h*64 + hd), f16
#pragma unroll
    for (int n = 0; n < 4; ++n)
#pragma unroll
        for (int r = 0; r < 8; ++r) {
            int row = i * 64 + w * 16 + r + 8 * half_id;
            int col = h * 64 + n * 16 + ln;
            attn[(size_t)row * 512 + col] = (_Float16)oacc[n][r];
        }
}

// --------------------------- output projection -----------------------------
// out[8192,512] f32 = attn_f16[8192,512] @ w_out_f16[512,512] + b_out
__global__ __launch_bounds__(128) void out_gemm_kernel(
    const _Float16* __restrict__ A, const _Float16* __restrict__ B,
    const float* __restrict__ bias, float* __restrict__ out) {
    __shared__ __align__(128) _Float16 sA[64 * 64];
    __shared__ __align__(128) _Float16 sBT[64 * 64];

    const int tid = threadIdx.x, lane = tid & 31, w = tid >> 5;
    const int row0 = blockIdx.y * 64, col0 = blockIdx.x * 64;
    const int half_id = lane >> 4, ln = lane & 15;

    v8f acc[4] = {};

    for (int ks = 0; ks < 512; ks += 64) {
        __syncthreads();
#if HAVE_ASYNC_LDS
        for (int t = tid; t < 512; t += 128) {
            int r = t >> 3, seg = t & 7;
            async_b128(&A[(size_t)(row0 + r) * 512 + ks + seg * 8],
                       &sA[r * 64 + seg * 8]);
        }
#else
        for (int t = tid; t < 512; t += 128) {
            int r = t >> 3, seg = t & 7;
            *(uint4*)&sA[r * 64 + seg * 8] =
                *(const uint4*)&A[(size_t)(row0 + r) * 512 + ks + seg * 8];
        }
#endif
        for (int t = tid; t < 512; t += 128) {
            int k = t >> 3, seg = t & 7;
            union { uint4 u; _Float16 h[8]; } cv;
            cv.u = *(const uint4*)&B[(size_t)(ks + k) * 512 + col0 + seg * 8];
#pragma unroll
            for (int e = 0; e < 8; ++e) sBT[(seg * 8 + e) * 64 + k] = cv.h[e];
        }
#if HAVE_ASYNC_LDS
        wait_async0();
#endif
        __syncthreads();

        v16h a0 = load_a_frag(sA, w * 16, 0, 64, lane);
        v16h a1 = load_a_frag(sA, w * 16, 32, 64, lane);
#pragma unroll
        for (int n = 0; n < 4; ++n) {
            v16h b0 = load_b_frag(sBT, n * 16, 0, 64, lane);
            v16h b1 = load_b_frag(sBT, n * 16, 32, 64, lane);
            acc[n] = WMMA_F16(a0, b0, acc[n]);
            acc[n] = WMMA_F16(a1, b1, acc[n]);
        }
    }

#pragma unroll
    for (int n = 0; n < 4; ++n)
#pragma unroll
        for (int r = 0; r < 8; ++r) {
            int row = row0 + w * 16 + r + 8 * half_id;
            int col = col0 + n * 16 + ln;
            out[(size_t)row * 512 + col] = acc[n][r] + bias[col];
        }
}

// ------------------------------- launcher ----------------------------------
extern "C" void kernel_launch(void* const* d_in, const int* in_sizes, int n_in,
                              void* d_out, int out_size, void* d_ws, size_t ws_size,
                              hipStream_t stream) {
    (void)in_sizes; (void)n_in; (void)out_size; (void)ws_size;
    const float* x     = (const float*)d_in[0];
    const float* w_qkv = (const float*)d_in[1];
    const float* b_qkv = (const float*)d_in[2];
    const float* w_out = (const float*)d_in[3];
    const float* b_out = (const float*)d_in[4];
    const float* w_r1  = (const float*)d_in[5];
    const float* b_r1  = (const float*)d_in[6];
    const float* w_r2  = (const float*)d_in[7];
    const float* b_r2  = (const float*)d_in[8];

    size_t off = 0;
    auto wsalloc = [&](size_t bytes) -> void* {
        void* p = (char*)d_ws + off;
        off += (bytes + 255) & ~(size_t)255;
        return p;
    };

    const int NX   = 8192 * 512;   // x / attn elements
    const int NWQ  = 512 * 1536;
    const int NWO  = 512 * 512;
    const int NQKV = 8 * 8192 * 64;

    _Float16* xa    = (_Float16*)wsalloc((size_t)NX * 2);   // x_f16, reused as attn_f16
    _Float16* wq16  = (_Float16*)wsalloc((size_t)NWQ * 2);
    _Float16* wo16  = (_Float16*)wsalloc((size_t)NWO * 2);
    _Float16* qb    = (_Float16*)wsalloc((size_t)NQKV * 2);
    _Float16* kb    = (_Float16*)wsalloc((size_t)NQKV * 2);
    _Float16* vb    = (_Float16*)wsalloc((size_t)NQKV * 2);
    float*    rep   = (float*)wsalloc((size_t)128 * 512 * 4);
    float*    keepv = (float*)wsalloc((size_t)128 * 4);

    cvt_f32_to_f16<<<(NX + 255) / 256, 256, 0, stream>>>(x, xa, NX);
    cvt_f32_to_f16<<<(NWQ + 255) / 256, 256, 0, stream>>>(w_qkv, wq16, NWQ);
    cvt_f32_to_f16<<<(NWO + 255) / 256, 256, 0, stream>>>(w_out, wo16, NWO);

    qkv_gemm_kernel<<<dim3(24, 128), 128, 0, stream>>>(xa, wq16, b_qkv, qb, kb, vb);

    rep_kernel<<<128, 512, 0, stream>>>(qb, rep);
    router_kernel<<<128, 128, 0, stream>>>(rep, w_r1, b_r1, w_r2, b_r2, keepv);

    attn_kernel<<<1024, 128, 0, stream>>>(qb, kb, vb, keepv, xa);  // attn overwrites xa

    out_gemm_kernel<<<dim3(8, 128), 128, 0, stream>>>(xa, wo16, b_out, (float*)d_out);
}
